// GNNChildDecoder_3951369912813
// MI455X (gfx1250) — matrix operand order repaired and checked by reading, not compiled, not run on hardware
//
#include <hip/hip_runtime.h>
#include <hip/hip_bf16.h>

// ---------------------------------------------------------------------------
// Sizes (fixed by the reference)
// ---------------------------------------------------------------------------
#define N 256
#define H 256
#define ITERS 3
// W_ne second dim = 3H+T = 769

typedef __attribute__((ext_vector_type(16))) _Float16 v16h;
typedef __attribute__((ext_vector_type(8)))  float    v8f;

union Frag16 { v16h v; int4 q[2]; };

// A-fragment (16x32 f16, ISA 7.12.2): lane<16 -> row M=lane, K = {0..7, 16..23};
// lane>=16 -> row M=lane-16, K = {8..15, 24..31}. Caller passes p = row + kbase + hi*8.
__device__ inline v16h load_frag_a(const _Float16* p) {
    Frag16 f;
    f.q[0] = *(const int4*)(p);        // K lo..lo+7
    f.q[1] = *(const int4*)(p + 16);   // K lo+16..lo+23
    return f.v;
}
// B-fragment (32x16 f16): lane<16 -> col N=lane, K = {0..7, 16..23}; lane>=16 -> col N=lane-16, K = {8..15, 24..31}.
// sB is stored as W3^T row-major [n][k]; caller passes p = colrow + kbase + hi*8.
__device__ inline v16h load_frag_b(const _Float16* p) {
    Frag16 f;
    f.q[0] = *(const int4*)(p);        // K lo..lo+7
    f.q[1] = *(const int4*)(p + 16);   // K lo+16..lo+23
    return f.v;
}

// gfx1250 async global->LDS copy, 16B per lane, tracked by ASYNCcnt.
// LDS address = low 32 bits of the generic pointer (ISA 10.2).
__device__ inline void async_copy_b128(unsigned ldsDst, const void* gSrc) {
    asm volatile("global_load_async_to_lds_b128 %0, %1, off"
                 :: "v"(ldsDst), "v"((unsigned long long)(uintptr_t)gSrc)
                 : "memory");
}
__device__ inline void wait_async0() {
    asm volatile("s_wait_asynccnt 0" ::: "memory");
}
__device__ inline unsigned lds_off(const void* p) {
    return (unsigned)(uintptr_t)p;     // addr[31:0] == LDS byte address
}

// ---------------------------------------------------------------------------
// Generic small linear: dst = (relu?) bias + src @ W.
// transpose==0: dst[i*lddst + h] ; transpose==1: dst[h*lddst + i]
// One block per row i, 256 threads (h). src row staged in LDS.
// ---------------------------------------------------------------------------
__global__ __launch_bounds__(256) void k_linear(
    const float* __restrict__ src, int ldsrc,
    const float* __restrict__ W,          // K x 256 row-major
    const float* __restrict__ bias,       // 256 or nullptr
    float* __restrict__ dst, int lddst,
    int K, int relu, int transpose)
{
    __shared__ float s[1024];
    const int i = blockIdx.x, h = threadIdx.x;
    for (int k = h; k < K; k += 256) s[k] = src[(size_t)i * ldsrc + k];
    __syncthreads();
    float acc = bias ? bias[h] : 0.0f;
    for (int k = 0; k < K; ++k) acc = fmaf(s[k], W[(size_t)k * 256 + h], acc);
    if (relu) acc = fmaxf(acc, 0.0f);
    if (transpose) dst[(size_t)h * lddst + i] = acc;
    else           dst[(size_t)i * lddst + h] = acc;
}

// exists[i] = (cf[i,:] . W_exists + b_exists) > 0 ? 1 : 0
__global__ __launch_bounds__(256) void k_exists(
    const float* __restrict__ cf, const float* __restrict__ Wex,
    const float* __restrict__ bex, float* __restrict__ exists)
{
    const int i = threadIdx.x;
    float acc = bex[0];
    for (int k = 0; k < H; ++k) acc = fmaf(cf[i * H + k], Wex[k], acc);
    exists[i] = (acc > 0.0f) ? 1.0f : 0.0f;
}

// W3T[it][n*256+k] = (f16) W_ne[it][(512+k)*256 + n]
__global__ __launch_bounds__(256) void k_w3t(
    const float* __restrict__ W_ne, _Float16* __restrict__ W3T)
{
    const int it = blockIdx.x, n = blockIdx.y, k = threadIdx.x;
    const float w = W_ne[(size_t)it * 769 * 256 + (size_t)(512 + k) * 256 + n];
    W3T[(size_t)it * 256 * 256 + (size_t)n * 256 + k] = (_Float16)w;
}

// cfbuf[:,0:256] = child_feats (row stride 1024 for the later concat)
__global__ __launch_bounds__(256) void k_copy_cf0(
    const float* __restrict__ src, float* __restrict__ dst)
{
    dst[(size_t)blockIdx.x * 1024 + threadIdx.x] = src[(size_t)blockIdx.x * 256 + threadIdx.x];
}

// ---------------------------------------------------------------------------
// Edge builder: one block per i; each wave owns 32 j rows (no block barriers
// in the loop; edge-logit dot reduced with intra-wave shuffles).
//   el = relu(A[i,h] + B[j,h])  -> EL16 (f16)
//   logit[i,j] = el . Wex_edge + b
//   maskf[i,j] = exists[i] & exists[j] & (logit > 0)
// ---------------------------------------------------------------------------
__global__ __launch_bounds__(256) void k_edge(
    const float* __restrict__ A, const float* __restrict__ Bm,
    const float* __restrict__ WexE, const float* __restrict__ bexE,
    const float* __restrict__ exists,
    _Float16* __restrict__ EL16, float* __restrict__ logit,
    float* __restrict__ maskf)
{
    __shared__ float sAi[256], sW[256];
    const int i = blockIdx.x, tid = threadIdx.x;
    const int wid = tid >> 5, lane = tid & 31;
    sAi[tid] = A[(size_t)i * 256 + tid];
    sW[tid]  = WexE[tid];
    __syncthreads();
    const float bex = bexE[0];
    const float exi = exists[i];
    for (int jj = 0; jj < 32; ++jj) {
        const int j = wid * 32 + jj;
        float p = 0.0f;
        #pragma unroll
        for (int s = 0; s < 8; ++s) {
            const int h = s * 32 + lane;
            float el = fmaxf(sAi[h] + Bm[(size_t)j * 256 + h], 0.0f);
            EL16[((size_t)(i * 256 + j)) * 256 + h] = (_Float16)el;
            p = fmaf(el, sW[h], p);
        }
        for (int off = 16; off > 0; off >>= 1) p += __shfl_xor(p, off, 32);
        if (lane == 0) {
            const float lg = p + bex;
            logit[(size_t)i * 256 + j] = lg;
            maskf[(size_t)i * 256 + j] =
                (lg > 0.0f && exi > 0.0f && exists[j] > 0.0f) ? 1.0f : 0.0f;
        }
    }
}

// ---------------------------------------------------------------------------
// Message-passing GEMM + fused epilogue + max-reduce (the WMMA kernel).
// One block per node i (256 blocks, 256 threads = 8 waves).
//   G[j,h] = EL16[i*256+j, :] @ W3            (256x256x256, f16 WMMA, f32 acc)
//   msg    = relu(G + U[i,h] + VT[h,j] + logit[i,j]*w4[h]) * maskf[i,j]
//   cf_next[i,h] = max_j msg[j,h]   (all values >= 0 -> int atomicMax trick)
// LDS: sB = W3^T f16 (128KB), double-buffered 64-row A chunks (2x32KB), 3KB misc.
// All global->LDS staging uses GLOBAL_LOAD_ASYNC_TO_LDS_B128 (ASYNCcnt),
// prefetching chunk jc+1 while chunk jc runs on the WMMA pipe.
// ---------------------------------------------------------------------------
#define KMSG_SMEM (131072 + 2 * 32768 + 1024 + 1024 + 1024)   // 199680 B

__global__ __launch_bounds__(256) void k_msg(
    const _Float16* __restrict__ EL16,
    const _Float16* __restrict__ W3T,   // [256][256] = W3^T, f16
    const float* __restrict__ U,        // [N,256] (includes b_ne)
    const float* __restrict__ VT,       // [256,N] = V^T (VT[h*256 + j])
    const float* __restrict__ logit,    // [N*N]
    const float* __restrict__ maskf,    // [N*N]
    const float* __restrict__ w4,       // [256] (row 768 of W_ne[it])
    float* __restrict__ dst, int lddst) // cfbuf + (it+1)*256, ld 1024
{
    extern __shared__ char smem[];
    _Float16* sB  = (_Float16*)smem;                       // 256*256 f16
    _Float16* sA0 = (_Float16*)(smem + 131072);            // 64*256 f16
    _Float16* sA1 = (_Float16*)(smem + 131072 + 32768);    // 64*256 f16
    float*    sU  = (float*)(smem + 131072 + 65536);       // 256 f32
    float*    sW4 = sU + 256;                              // 256 f32
    int*      sMax = (int*)(sW4 + 256);                    // 256 i32

    const int i   = blockIdx.x;
    const int tid = threadIdx.x;
    const int wid = tid >> 5, lane = tid & 31;
    const int waveM = wid & 1, waveN = wid >> 1;           // 2 x 4 wave grid
    const int lr = lane & 15, hi = lane >> 4;

    const unsigned aOff[2] = { lds_off(sA0), lds_off(sA1) };

    // Async-stage full W3^T (8192 x b128) and A chunk 0 (2048 x b128).
    {
        const unsigned bOff = lds_off(sB);
        const char* g = (const char*)W3T;
        for (int t = tid; t < 8192; t += 256)
            async_copy_b128(bOff + t * 16, g + (size_t)t * 16);
    }
    {
        const char* g = (const char*)(EL16 + ((size_t)(i * 256)) * 256);
        for (int t = tid; t < 2048; t += 256)
            async_copy_b128(aOff[0] + t * 16, g + (size_t)t * 16);
    }
    sU[tid]   = U[(size_t)i * 256 + tid];
    sW4[tid]  = w4[tid];
    sMax[tid] = 0;                                         // 0.0f bit pattern

    for (int jc = 0; jc < 4; ++jc) {                       // 64 rows per chunk
        wait_async0();                                     // own async slice done
        __syncthreads();                                   // everyone's slice done
        if (jc < 3) {                                      // prefetch next chunk
            const char* g = (const char*)(EL16 + ((size_t)(i * 256 + (jc + 1) * 64)) * 256);
            const unsigned dstOff = aOff[(jc + 1) & 1];
            for (int t = tid; t < 2048; t += 256)
                async_copy_b128(dstOff + t * 16, g + (size_t)t * 16);
        }
        const _Float16* sA = (jc & 1) ? sA1 : sA0;

        v8f acc[2][4];
        v8f zero = {};
        #pragma unroll
        for (int mt = 0; mt < 2; ++mt)
            #pragma unroll
            for (int nt = 0; nt < 4; ++nt) acc[mt][nt] = zero;

        for (int kc = 0; kc < 256; kc += 32) {
            v16h a[2], b[4];
            #pragma unroll
            for (int mt = 0; mt < 2; ++mt) {
                const int row = waveM * 32 + mt * 16 + lr;
                a[mt] = load_frag_a(sA + row * 256 + kc + hi * 8);
            }
            #pragma unroll
            for (int nt = 0; nt < 4; ++nt) {
                const int col = waveN * 64 + nt * 16 + lr;
                b[nt] = load_frag_b(sB + col * 256 + kc + hi * 8);
            }
            #pragma unroll
            for (int mt = 0; mt < 2; ++mt)
                #pragma unroll
                for (int nt = 0; nt < 4; ++nt)
                    acc[mt][nt] = __builtin_amdgcn_wmma_f32_16x16x32_f16(
                        false, a[mt], false, b[nt], (short)0, acc[mt][nt],
                        false, false);
        }

        // Fused epilogue + row-max over j for this chunk (vectorized loads).
        #pragma unroll
        for (int mt = 0; mt < 2; ++mt) {
            const int jbase = jc * 64 + waveM * 32 + mt * 16 + hi * 8;
            const size_t r0 = (size_t)i * 256 + jbase;
            const float4 lgA = *(const float4*)(logit + r0);
            const float4 lgB = *(const float4*)(logit + r0 + 4);
            const float4 mkA = *(const float4*)(maskf + r0);
            const float4 mkB = *(const float4*)(maskf + r0 + 4);
            const float lg[8] = {lgA.x, lgA.y, lgA.z, lgA.w, lgB.x, lgB.y, lgB.z, lgB.w};
            const float mk[8] = {mkA.x, mkA.y, mkA.z, mkA.w, mkB.x, mkB.y, mkB.z, mkB.w};
            #pragma unroll
            for (int nt = 0; nt < 4; ++nt) {
                const int n = waveN * 64 + nt * 16 + lr;
                const float4 vA = *(const float4*)(VT + (size_t)n * 256 + jbase);
                const float4 vB = *(const float4*)(VT + (size_t)n * 256 + jbase + 4);
                const float vj[8] = {vA.x, vA.y, vA.z, vA.w, vB.x, vB.y, vB.z, vB.w};
                const float un = sU[n], w4n = sW4[n];
                float vmax = 0.0f;
                #pragma unroll
                for (int v = 0; v < 8; ++v) {
                    float val = acc[mt][nt][v] + un + vj[v] + lg[v] * w4n;
                    vmax = fmaxf(vmax, fmaxf(val, 0.0f) * mk[v]);
                }
                atomicMax(&sMax[n], __float_as_int(vmax));
            }
        }
    }
    __syncthreads();
    dst[(size_t)i * lddst + tid] = __int_as_float(sMax[tid]);
}

// ---------------------------------------------------------------------------
// Host launch
// ---------------------------------------------------------------------------
extern "C" void kernel_launch(void* const* d_in, const int* in_sizes, int n_in,
                              void* d_out, int out_size, void* d_ws, size_t ws_size,
                              hipStream_t stream) {
    (void)in_sizes; (void)n_in; (void)out_size; (void)ws_size;

    const float* cf        = (const float*)d_in[0];   // [1,N,H]
    const float* W_exists  = (const float*)d_in[1];   // [H,1]
    const float* b_exists  = (const float*)d_in[2];   // [1]
    const float* W_edge    = (const float*)d_in[3];   // [2H,H]
    const float* b_edge    = (const float*)d_in[4];   // [H]
    const float* W_edge_ex = (const float*)d_in[5];   // [T=1,H]
    const float* b_edge_ex = (const float*)d_in[6];   // [1]
    const float* W_ne      = (const float*)d_in[7];   // [3,769,H]
    const float* b_ne      = (const float*)d_in[8];   // [3,H]
    const float* W_child   = (const float*)d_in[9];   // [4H,H]
    const float* b_child   = (const float*)d_in[10];  // [H]
    const float* W_child2  = (const float*)d_in[11];  // [H,H]
    const float* b_child2  = (const float*)d_in[12];  // [H]
    float* out = (float*)d_out;

    // Workspace carve-up (all offsets 256B-aligned)
    char* ws = (char*)d_ws;
    float* cfbuf  = (float*)ws;             ws += (size_t)N * 1024 * 4;  // [N][4][H] concat layout
    float* Abuf   = (float*)ws;             ws += (size_t)N * H * 4;
    float* Bbuf   = (float*)ws;             ws += (size_t)N * H * 4;
    float* Ubuf   = (float*)ws;             ws += (size_t)N * H * 4;
    float* VTbuf  = (float*)ws;             ws += (size_t)N * H * 4;     // V transposed [H][N]
    float* logitb = (float*)ws;             ws += (size_t)N * N * 4;
    float* maskb  = (float*)ws;             ws += (size_t)N * N * 4;
    float* tmp1   = (float*)ws;             ws += (size_t)N * H * 4;
    float* exists = (float*)ws;             ws += 4096;
    _Float16* W3T  = (_Float16*)ws;         ws += (size_t)ITERS * H * H * 2;
    _Float16* EL16 = (_Float16*)ws;         ws += (size_t)N * N * H * 2;   // 32 MB

    // Allow >64KB dynamic LDS for the WMMA kernel (CDNA5: up to 320KB/WG)
    (void)hipFuncSetAttribute(reinterpret_cast<const void*>(k_msg),
                              hipFuncAttributeMaxDynamicSharedMemorySize,
                              KMSG_SMEM);

    // Phase A: node-level linears
    k_exists<<<1, 256, 0, stream>>>(cf, W_exists, b_exists, exists);
    // A = cf @ W_top + b_edge ; B = cf @ W_bot
    k_linear<<<N, 256, 0, stream>>>(cf, H, W_edge,             b_edge,  Abuf, H, H, 0, 0);
    k_linear<<<N, 256, 0, stream>>>(cf, H, W_edge + 256 * 256, nullptr, Bbuf, H, H, 0, 0);
    k_w3t<<<dim3(ITERS, 256), 256, 0, stream>>>(W_ne, W3T);
    k_copy_cf0<<<N, 256, 0, stream>>>(cf, cfbuf);

    // Phase B: edge latents (f16) + logits + mask
    k_edge<<<N, 256, 0, stream>>>(Abuf, Bbuf, W_edge_ex, b_edge_ex, exists,
                                  EL16, logitb, maskb);

    // Phase C: 3 message-passing iterations
    for (int it = 0; it < ITERS; ++it) {
        const float* Wit = W_ne + (size_t)it * 769 * 256;
        const float* cfi = cfbuf + (size_t)it * 256;      // row stride 1024
        // U = cf_it @ W1 + b_ne[it] ; V^T = (cf_it @ W2)^T
        k_linear<<<N, 256, 0, stream>>>(cfi, 1024, Wit,             b_ne + (size_t)it * 256, Ubuf,  H, H, 0, 0);
        k_linear<<<N, 256, 0, stream>>>(cfi, 1024, Wit + 256 * 256, nullptr,                 VTbuf, N, H, 0, 1);
        k_msg<<<N, 256, KMSG_SMEM, stream>>>(
            EL16, W3T + (size_t)it * 256 * 256, Ubuf, VTbuf, logitb, maskb,
            Wit + (size_t)768 * 256,                 // w4 (type-logit column)
            cfbuf + (size_t)(it + 1) * 256, 1024);
    }

    // Phase D: final MLPs on concat(cf0..cf3)
    k_linear<<<N, 256, 0, stream>>>(cfbuf, 1024, W_child,  b_child,  tmp1, H, 1024, 1, 0);
    k_linear<<<N, 256, 0, stream>>>(tmp1,  H,    W_child2, b_child2, out,  H, H,    1, 0);
}
